// Relation_block_batch_1984274890946
// MI455X (gfx1250) — compile-verified, matrix operand order checked by reading, not run on hardware
//
#include <hip/hip_runtime.h>
#include <hip/hip_bf16.h>

// ---------------------------------------------------------------------------
// out[n,d] = pk[n,d] + b[d] + max_m ( O[n,m,:] . Wo[d,:] )
// pk[n,d]  = P[n,:] . Wp[d,:]
// N=2048, M=32, C=1024.  Split-bf16 (hi/lo) WMMA GEMM, f32 accumulate.
// ---------------------------------------------------------------------------

typedef __attribute__((ext_vector_type(16))) __bf16 v16bf;
typedef __attribute__((ext_vector_type(8)))  __bf16 v8bf;
typedef __attribute__((ext_vector_type(4)))  __bf16 v4bf;
typedef __attribute__((ext_vector_type(8)))  float  v8f;

#define BM       128              // rows per block (= 4 n-groups of 32 m-rows)
#define BN       128              // output columns per block
#define KDEPTH   1024             // contraction length (C)
#define KC       32               // K chunk per LDS stage (one bf16 wmma K)
#define NKC      (KDEPTH / KC)    // 32 stages
#define LDK      40               // padded LDS row stride (conflict-free, 16B aligned)
#define PLANE    (BM * LDK)       // elements per LDS plane
#define NTHREADS 256              // 8 waves (wave32)

// Load one 16x32 bf16 WMMA fragment from an LDS plane stored row-major [16][LDK].
// Per ISA layout: lanes 0-15 hold row = lane, K = {0..7, 16..23};
// lanes 16-31 hold row = lane-16, K = {8..15, 24..31}.  Two 16B ds reads.
__device__ __forceinline__ v16bf frag_ld(const __bf16* rowbase, int lane) {
    int r  = lane & 15;
    int kb = (lane >> 4) << 3;                 // 0 or 8
    const __bf16* p = rowbase + r * LDK + kb;
    v8bf lo = *(const v8bf*)(p);               // K = kb .. kb+7
    v8bf hi = *(const v8bf*)(p + 16);          // K = kb+16 .. kb+23
    return __builtin_shufflevector(lo, hi, 0,1,2,3,4,5,6,7,8,9,10,11,12,13,14,15);
}

// f32 -> (hi bf16, lo bf16) split; store 4 elements (8B) to each plane.
__device__ __forceinline__ void cvt_store(__bf16* hiP, __bf16* loP, float4 v) {
    __bf16 h0 = (__bf16)v.x, h1 = (__bf16)v.y, h2 = (__bf16)v.z, h3 = (__bf16)v.w;
    __bf16 l0 = (__bf16)(v.x - (float)h0);
    __bf16 l1 = (__bf16)(v.y - (float)h1);
    __bf16 l2 = (__bf16)(v.z - (float)h2);
    __bf16 l3 = (__bf16)(v.w - (float)h3);
    v4bf h; h[0] = h0; h[1] = h1; h[2] = h2; h[3] = h3;
    v4bf l; l[0] = l0; l[1] = l1; l[2] = l2; l[3] = l3;
    *(v4bf*)hiP = h;
    *(v4bf*)loP = l;
}

// Core: acc[2][4] (32 rows x 64 cols per wave) += A(BMxK) * B(BNxK)^T using
// 3-product split-bf16 wmma.  A row stride = lda floats, B row stride = ldb.
__device__ __forceinline__ void gemm_core(const float* __restrict__ A, size_t lda,
                                          const float* __restrict__ B, size_t ldb,
                                          __bf16* smem, v8f acc[2][4]) {
    const int tid  = threadIdx.x;
    const int lane = tid & 31;
    const int wave = tid >> 5;
    const int wr   = wave & 3;     // 32-row stripe (= one n group in kernel 2)
    const int wc   = wave >> 2;    // 64-col stripe

    __bf16* Ah = smem;
    __bf16* Al = smem + PLANE;
    __bf16* Bh = smem + 2 * PLANE;
    __bf16* Bl = smem + 3 * PLANE;

    // Each thread loads 4 float4 of A and 4 of B per chunk (128 rows x 32 k).
    int rr[4], kk[4];
#pragma unroll
    for (int i = 0; i < 4; ++i) {
        int flat = i * NTHREADS + tid;
        rr[i] = flat >> 3;         // row within chunk
        kk[i] = (flat & 7) << 2;   // k offset (float4 granularity)
    }

    float4 aReg[4], bReg[4];
#pragma unroll
    for (int i = 0; i < 4; ++i) {  // prefetch chunk 0
        aReg[i] = *(const float4*)(A + (size_t)rr[i] * lda + kk[i]);
        bReg[i] = *(const float4*)(B + (size_t)rr[i] * ldb + kk[i]);
    }

    for (int kc = 0; kc < NKC; ++kc) {
        __syncthreads();           // previous stage's ds_reads done
#pragma unroll
        for (int i = 0; i < 4; ++i) {
            cvt_store(Ah + rr[i] * LDK + kk[i], Al + rr[i] * LDK + kk[i], aReg[i]);
            cvt_store(Bh + rr[i] * LDK + kk[i], Bl + rr[i] * LDK + kk[i], bReg[i]);
        }
        __syncthreads();           // stage visible to all waves

        if (kc + 1 < NKC) {        // prefetch next chunk: loads in flight during wmma
            const float* An = A + (size_t)(kc + 1) * KC;
            const float* Bn = B + (size_t)(kc + 1) * KC;
#pragma unroll
            for (int i = 0; i < 4; ++i) {
                aReg[i] = *(const float4*)(An + (size_t)rr[i] * lda + kk[i]);
                bReg[i] = *(const float4*)(Bn + (size_t)rr[i] * ldb + kk[i]);
            }
        }

        v16bf aH[2], aL[2];
#pragma unroll
        for (int tr = 0; tr < 2; ++tr) {
            aH[tr] = frag_ld(Ah + (wr * 32 + tr * 16) * LDK, lane);
            aL[tr] = frag_ld(Al + (wr * 32 + tr * 16) * LDK, lane);
        }
#pragma unroll
        for (int tc = 0; tc < 4; ++tc) {
            v16bf bHf = frag_ld(Bh + (wc * 64 + tc * 16) * LDK, lane);
            v16bf bLf = frag_ld(Bl + (wc * 64 + tc * 16) * LDK, lane);
#pragma unroll
            for (int tr = 0; tr < 2; ++tr) {
                acc[tr][tc] = __builtin_amdgcn_wmma_f32_16x16x32_bf16(
                    false, aH[tr], false, bHf, (short)0, acc[tr][tc], false, false);
                acc[tr][tc] = __builtin_amdgcn_wmma_f32_16x16x32_bf16(
                    false, aH[tr], false, bLf, (short)0, acc[tr][tc], false, false);
                acc[tr][tc] = __builtin_amdgcn_wmma_f32_16x16x32_bf16(
                    false, aL[tr], false, bHf, (short)0, acc[tr][tc], false, false);
            }
        }
    }
}

// Kernel 1: pk = P(2048x1024) @ Wp^T  -> d_out (2048x1024)
__global__ __launch_bounds__(NTHREADS) void pk_gemm_kernel(
    const float* __restrict__ P, const float* __restrict__ Wp, float* __restrict__ Out) {
    __shared__ __align__(16) __bf16 smem[4 * PLANE];
    v8f acc[2][4] = {};
    const int aRow0 = blockIdx.y * BM;   // n-block
    const int dCol0 = blockIdx.x * BN;   // d-block
    gemm_core(P + (size_t)aRow0 * KDEPTH, KDEPTH,
              Wp + (size_t)dCol0 * (2 * KDEPTH), 2 * KDEPTH, smem, acc);

    const int lane  = threadIdx.x & 31;
    const int wave  = threadIdx.x >> 5;
    const int wr    = wave & 3, wc = wave >> 2;
    const int rHalf = (lane >> 4) << 3;            // VGPR v holds rows v / v+8
    const int col   = dCol0 + wc * 64 + (lane & 15);
#pragma unroll
    for (int tr = 0; tr < 2; ++tr)
#pragma unroll
        for (int tc = 0; tc < 4; ++tc)
#pragma unroll
            for (int v = 0; v < 8; ++v) {
                int row = aRow0 + wr * 32 + tr * 16 + rHalf + v;
                Out[(size_t)row * 1024 + col + tc * 16] = acc[tr][tc][v];
            }
}

// Kernel 2: out[n,d] = pk[n,d] (already in Out) + b[d] + max_m (O @ Wo^T)
__global__ __launch_bounds__(NTHREADS) void relmax_kernel(
    const float* __restrict__ O, const float* __restrict__ Wo,
    const float* __restrict__ bias, float* __restrict__ Out) {
    __shared__ __align__(16) __bf16 smem[4 * PLANE];
    v8f acc[2][4] = {};
    const int aRow0 = blockIdx.y * BM;   // 128 rows = 4 full n groups (m=0..31 each)
    const int dCol0 = blockIdx.x * BN;
    gemm_core(O + (size_t)aRow0 * KDEPTH, KDEPTH,
              Wo + (size_t)dCol0 * (2 * KDEPTH), 2 * KDEPTH, smem, acc);

    const int lane = threadIdx.x & 31;
    const int wave = threadIdx.x >> 5;
    const int wr   = wave & 3, wc = wave >> 2;
    const int n    = blockIdx.y * 4 + wr;          // this wave's 32 rows = one n
#pragma unroll
    for (int tc = 0; tc < 4; ++tc) {
        float m = acc[0][tc][0];
#pragma unroll
        for (int tr = 0; tr < 2; ++tr)
#pragma unroll
            for (int v = 0; v < 8; ++v) m = fmaxf(m, acc[tr][tc][v]);
        // lanes L / L+16 hold complementary row halves of the same column
        m = fmaxf(m, __shfl_xor(m, 16, 32));
        if (lane < 16) {
            int d = dCol0 + wc * 64 + tc * 16 + lane;
            size_t idx = (size_t)n * 1024 + d;
            Out[idx] = Out[idx] + bias[d] + m;     // pk + b + max_m ok
        }
    }
}

extern "C" void kernel_launch(void* const* d_in, const int* in_sizes, int n_in,
                              void* d_out, int out_size, void* d_ws, size_t ws_size,
                              hipStream_t stream) {
    (void)in_sizes; (void)n_in; (void)out_size; (void)d_ws; (void)ws_size;
    const float* P  = (const float*)d_in[0];        // (2048,1024,1,1)
    const float* O  = (const float*)d_in[1];        // (2048,32,1024,1,1)
    const float* W  = (const float*)d_in[5];        // (1024, 2048): [:, :1024]=Wp, [:,1024:]=Wo
    const float* b  = (const float*)d_in[6];        // (1024,)
    float*       out = (float*)d_out;               // (2048,1024,1,1)

    dim3 blk(NTHREADS);
    // pk into d_out: 2048/128 n-blocks x 1024/128 d-blocks (x = d-block for L2 reuse)
    pk_gemm_kernel<<<dim3(KDEPTH / BN, 2048 / BM), blk, 0, stream>>>(P, W, out);
    // fused ok-GEMM + max-over-m + (pk + b) epilogue: 65536/128 row-blocks
    relmax_kernel<<<dim3(KDEPTH / BN, 65536 / BM), blk, 0, stream>>>(O, W + 1024, b, out);
}
